// ModalityUntiedFeedForward_1477468749957
// MI455X (gfx1250) — matrix operand order repaired and checked by reading, not compiled, not run on hardware
//
#include <hip/hip_runtime.h>

// ---------------- problem constants (match reference) ----------------
#define DIMD 2048      // model dim (K of gemm1/3, N of gemm2)
#define HIDD 3840      // hidden dim
#define NEXP 4         // experts / modalities
#define NTOK 8192      // BS*SEQ tokens
#define EPSR 1e-5f

#define MT   32        // tokens per block tile
#define HCHK 256       // H chunk per iteration
#define NHCH (HIDD / HCHK)   // 15
#define XSTRIDE (DIMD + 8)   // bf16 elems, row stride for x tile in LDS
#define HSTRIDE (HCHK + 8)   // bf16 elems, row stride for h chunk in LDS

typedef __attribute__((ext_vector_type(16))) __bf16 v16bf;
typedef __attribute__((ext_vector_type(8)))  float  v8f;

// ---------------- small helpers ----------------
static __device__ inline v8f zero8() {
  v8f z;
#pragma unroll
  for (int r = 0; r < 8; ++r) z[r] = 0.0f;
  return z;
}

static __device__ inline v8f wmma_bf16(v16bf a, v16bf b, v8f c) {
  // (neg_a, A, neg_b, B, c_mod, C, reuse_a, reuse_b)
  return __builtin_amdgcn_wmma_f32_16x16x32_bf16(false, a, false, b, (short)0, c,
                                                 false, false);
}

// silu(g) = g * sigmoid(g), with hardware-rate v_rcp_f32 (no IEEE div expansion)
static __device__ inline float fast_silu(float g) {
  return g * __builtin_amdgcn_rcpf(1.0f + __expf(-g));
}

// A-fragment (16-bit 16x32): lane holds K = 32k+16*(l/16)+{0..15}
// p points at &row[32k + 16*lhi]; both 16B groups are 16B aligned.
static __device__ inline v16bf load_afrag(const __bf16* p) {
  union { uint4 q[2]; v16bf v; } u;
  u.q[0] = *(const uint4*)(p);
  u.q[1] = *(const uint4*)(p + 8);
  return u.v;
}

// B-fragment: 16 consecutive elements along K for this lane's column.
static __device__ inline v16bf load_bfrag(const __bf16* p) {
  union { uint4 q[2]; v16bf v; } u;
  u.q[0] = ((const uint4*)p)[0];
  u.q[1] = ((const uint4*)p)[1];
  return u.v;
}
static __device__ inline v16bf load_bfrag(const float* p) {
  union { float4 f[4]; float s[16]; } t;
  t.f[0] = ((const float4*)p)[0];
  t.f[1] = ((const float4*)p)[1];
  t.f[2] = ((const float4*)p)[2];
  t.f[3] = ((const float4*)p)[3];
  v16bf v;
#pragma unroll
  for (int j = 0; j < 16; ++j) v[j] = (__bf16)t.s[j];
  return v;
}

// ---------------- routing kernels ----------------
__global__ void zero_counts_kernel(int* counts) {
  if (threadIdx.x < NEXP) counts[threadIdx.x] = 0;
}

__global__ void route_kernel(const int* __restrict__ ids, int* __restrict__ counts,
                             int* __restrict__ idx) {
  int t = blockIdx.x * blockDim.x + threadIdx.x;
  if (t < NTOK) {
    int e = ids[t];
    int p = atomicAdd(&counts[e], 1);
    idx[(size_t)e * NTOK + p] = t;
  }
}

// ---------------- f32 -> bf16 weight conversion ----------------
__global__ void convert_kernel(const float* __restrict__ src,
                               __bf16* __restrict__ dst, size_t n4) {
  size_t i = (size_t)blockIdx.x * blockDim.x + threadIdx.x;
  size_t stride = (size_t)gridDim.x * blockDim.x;
  for (; i < n4; i += stride) {
    float4 v = ((const float4*)src)[i];
    union { __bf16 h[4]; uint2 u; } o;
    o.h[0] = (__bf16)v.x; o.h[1] = (__bf16)v.y;
    o.h[2] = (__bf16)v.z; o.h[3] = (__bf16)v.w;
    ((uint2*)dst)[i] = o.u;
  }
}

// ---------------- fused routed SwiGLU-MLP + RMSNorm ----------------
template <typename WT>
__global__ void __launch_bounds__(512, 1)
ffn_kernel(const float* __restrict__ x, const WT* __restrict__ w1,
           const WT* __restrict__ w2, const WT* __restrict__ w3,
           const float* __restrict__ norm_w, const int* __restrict__ counts,
           const int* __restrict__ idx, float* __restrict__ out) {
  const int e = blockIdx.y;
  const int cnt = counts[e];
  const int base = blockIdx.x * MT;
  if (base >= cnt) return;
  const int nrows = min(MT, cnt - base);

  __shared__ int   tok[MT];
  __shared__ float rowss[MT];
  __shared__ float rsig[MT];
  __shared__ __align__(16) __bf16 xs[MT * XSTRIDE];   // 131.6 KB
  __shared__ __align__(16) __bf16 hch[MT * HSTRIDE];  // 16.9 KB

  const int tid = threadIdx.x;
  if (tid < MT) {
    tok[tid] = idx[(size_t)e * NTOK + base + (tid < nrows ? tid : 0)];
    rowss[tid] = 0.0f;
  }
  __syncthreads();

  // Stage x tile (gathered tokens) into LDS as bf16, coalesced.
  for (int r = 0; r < MT; ++r) {
    const float* xr = x + (size_t)tok[r] * DIMD;
    for (int c = tid; c < DIMD; c += 512)
      xs[r * XSTRIDE + c] = (__bf16)xr[c];
  }
  __syncthreads();

  const int w    = tid >> 5;    // wave 0..15
  const int lane = tid & 31;
  const int lrow = lane & 15;   // M (A) or N (B/C)
  const int lhi  = lane >> 4;   // half-wave select

  const WT* W1e = w1 + (size_t)e * HIDD * DIMD;
  const WT* W3e = w3 + (size_t)e * HIDD * DIMD;
  const WT* W2e = w2 + (size_t)e * DIMD * HIDD;

  // Output accumulators: wave owns D-cols {16*(w+16i)+lrow}, rows all 32.
  v8f accY[2][8];
#pragma unroll
  for (int m = 0; m < 2; ++m)
#pragma unroll
    for (int i = 0; i < 8; ++i) accY[m][i] = zero8();

  for (int hc = 0; hc < NHCH; ++hc) {
    // ---- Phase A: h chunk = silu(x W1^T) * (x W3^T), wave w -> n=w ----
    {
      const int n = w;
      const int hrow = hc * HCHK + n * 16 + lrow;      // W1/W3 row (B column)
      const WT* bp1 = W1e + (size_t)hrow * DIMD;
      const WT* bp3 = W3e + (size_t)hrow * DIMD;
      v8f a1[2] = {zero8(), zero8()};
      v8f a3[2] = {zero8(), zero8()};
#pragma unroll 2
      for (int k = 0; k < DIMD / 32; ++k) {
        const int kb = k * 32 + 16 * lhi;
        v16bf B1 = load_bfrag(bp1 + kb);
        v16bf B3 = load_bfrag(bp3 + kb);
        v16bf A0 = load_afrag(&xs[(0  + lrow) * XSTRIDE + k * 32 + 16 * lhi]);
        v16bf A1 = load_afrag(&xs[(16 + lrow) * XSTRIDE + k * 32 + 16 * lhi]);
        a1[0] = wmma_bf16(A0, B1, a1[0]);
        a3[0] = wmma_bf16(A0, B3, a3[0]);
        a1[1] = wmma_bf16(A1, B1, a1[1]);
        a3[1] = wmma_bf16(A1, B3, a3[1]);
      }
#pragma unroll
      for (int m = 0; m < 2; ++m)
#pragma unroll
        for (int r = 0; r < 8; ++r) {
          float hval = fast_silu(a1[m][r]) * a3[m][r];
          int row = 16 * m + 8 * lhi + r;
          hch[row * HSTRIDE + n * 16 + lrow] = (__bf16)hval;
        }
    }
    __syncthreads();

    // ---- Phase B: y += h_chunk @ W2_chunk^T (acc in registers) ----
#pragma unroll 2
    for (int k = 0; k < HCHK / 32; ++k) {
      v16bf A0 = load_afrag(&hch[(0  + lrow) * HSTRIDE + k * 32 + 16 * lhi]);
      v16bf A1 = load_afrag(&hch[(16 + lrow) * HSTRIDE + k * 32 + 16 * lhi]);
#pragma unroll
      for (int i = 0; i < 8; ++i) {
        const int drow = (w + 16 * i) * 16 + lrow;     // W2 row (B column)
        const WT* bp2 =
            W2e + (size_t)drow * HIDD + hc * HCHK + k * 32 + 16 * lhi;
        v16bf B = load_bfrag(bp2);
        accY[0][i] = wmma_bf16(A0, B, accY[0][i]);
        accY[1][i] = wmma_bf16(A1, B, accY[1][i]);
      }
    }
    __syncthreads();
  }

  // ---- RMSNorm: per-row sum of squares via LDS atomics ----
#pragma unroll
  for (int m = 0; m < 2; ++m)
#pragma unroll
    for (int r = 0; r < 8; ++r) {
      float s = 0.0f;
#pragma unroll
      for (int i = 0; i < 8; ++i) {
        float v = accY[m][i][r];
        s += v * v;
      }
      atomicAdd(&rowss[16 * m + 8 * lhi + r], s);
    }
  __syncthreads();
  if (tid < MT)
    rsig[tid] = __builtin_amdgcn_rsqf(rowss[tid] * (1.0f / DIMD) + EPSR);
  __syncthreads();

  // ---- scale + scatter-write to each token's row ----
  const float* nwe = norm_w + (size_t)e * DIMD;
#pragma unroll
  for (int i = 0; i < 8; ++i) {
    const int col = (w + 16 * i) * 16 + lrow;
    const float nwv = nwe[col];
#pragma unroll
    for (int m = 0; m < 2; ++m)
#pragma unroll
      for (int r = 0; r < 8; ++r) {
        const int row = 16 * m + 8 * lhi + r;
        if (row < nrows)
          out[(size_t)tok[row] * DIMD + col] = accY[m][i][r] * rsig[row] * nwv;
      }
  }
}

// ---------------- host launcher ----------------
extern "C" void kernel_launch(void* const* d_in, const int* in_sizes, int n_in,
                              void* d_out, int out_size, void* d_ws,
                              size_t ws_size, hipStream_t stream) {
  (void)in_sizes; (void)n_in; (void)out_size;
  const float* x      = (const float*)d_in[0];
  const float* w1     = (const float*)d_in[1];
  const float* w2     = (const float*)d_in[2];
  const float* w3     = (const float*)d_in[3];
  const float* norm_w = (const float*)d_in[4];
  const int*   ids    = (const int*)d_in[5];
  float* out = (float*)d_out;

  int* counts = (int*)d_ws;
  int* idx    = (int*)((char*)d_ws + 256);

  zero_counts_kernel<<<1, 32, 0, stream>>>(counts);
  route_kernel<<<NTOK / 256, 256, 0, stream>>>(ids, counts, idx);

  const size_t EHD = (size_t)NEXP * HIDD * DIMD;          // elems per weight
  const size_t woff = (size_t)1 << 20;                    // 1 MB for routing
  const size_t ws_need = woff + 3 * EHD * sizeof(__bf16); // ~181 MB

  dim3 grid((NTOK + MT - 1) / MT, NEXP);

  if (ws_size >= ws_need) {
    __bf16* bw1 = (__bf16*)((char*)d_ws + woff);
    __bf16* bw3 = bw1 + EHD;
    __bf16* bw2 = bw3 + EHD;
    convert_kernel<<<2048, 256, 0, stream>>>(w1, bw1, EHD / 4);
    convert_kernel<<<2048, 256, 0, stream>>>(w3, bw3, EHD / 4);
    convert_kernel<<<2048, 256, 0, stream>>>(w2, bw2, EHD / 4);
    ffn_kernel<__bf16><<<grid, 512, 0, stream>>>(x, bw1, bw2, bw3, norm_w,
                                                 counts, idx, out);
  } else {
    ffn_kernel<float><<<grid, 512, 0, stream>>>(x, w1, w2, w3, norm_w, counts,
                                                idx, out);
  }
}
